// LSTM_55825984914080
// MI455X (gfx1250) — matrix operand order repaired and checked by reading, not compiled, not run on hardware
//
#include <hip/hip_runtime.h>
#include <hip/hip_bf16.h>
#include <math.h>

typedef __bf16 bf16;
typedef __attribute__((ext_vector_type(16))) __bf16 v16bf;
typedef __attribute__((ext_vector_type(8)))  __bf16 v8bf;
typedef __attribute__((ext_vector_type(8)))  float  v8f;
typedef __attribute__((ext_vector_type(4)))  unsigned int v4u;
typedef __attribute__((ext_vector_type(8)))  int v8i;
typedef __attribute__((ext_vector_type(4)))  int v4i;

#define B_      128
#define S_      512
#define D_      128
#define H_      1024
#define O_      64
#define IN_RAW  257
#define IN_PAD  288     // pad K to multiple of 32 for 16x16x32 WMMA
#define G4H     4096    // 4*H

__device__ __forceinline__ float sigmoidf_(float x) { return 1.0f / (1.0f + __expf(-x)); }

// ---------------- prep kernels ----------------

// xtb[s][b][i] = concat(x, mask, ti)[b][s][i], bf16, zero-padded to IN_PAD
__global__ void prep_xtb(const float* __restrict__ x, const float* __restrict__ mask,
                         const float* __restrict__ ti, bf16* __restrict__ xtb) {
    size_t total = (size_t)S_ * B_ * IN_PAD;
    for (size_t idx = (size_t)blockIdx.x * blockDim.x + threadIdx.x; idx < total;
         idx += (size_t)gridDim.x * blockDim.x) {
        int i = (int)(idx % IN_PAD);
        size_t sb = idx / IN_PAD;
        int b = (int)(sb % B_);
        int s = (int)(sb / B_);
        float v;
        if (i < D_)           v = x[((size_t)b * S_ + s) * D_ + i];
        else if (i < 2 * D_)  v = mask[((size_t)b * S_ + s) * D_ + (i - D_)];
        else if (i == 2 * D_) v = ti[(size_t)b * S_ + s];
        else                  v = 0.0f;
        xtb[idx] = (bf16)v;
    }
}

// W_ih [4096][257] fp32 -> [4096][288] bf16 (zero pad)
__global__ void prep_wih(const float* __restrict__ W, bf16* __restrict__ Wb) {
    size_t total = (size_t)G4H * IN_PAD;
    for (size_t idx = (size_t)blockIdx.x * blockDim.x + threadIdx.x; idx < total;
         idx += (size_t)gridDim.x * blockDim.x) {
        int i = (int)(idx % IN_PAD);
        int g = (int)(idx / IN_PAD);
        Wb[idx] = (bf16)((i < IN_RAW) ? W[(size_t)g * IN_RAW + i] : 0.0f);
    }
}

__global__ void prep_f2b(const float* __restrict__ src, bf16* __restrict__ dst, size_t n) {
    for (size_t idx = (size_t)blockIdx.x * blockDim.x + threadIdx.x; idx < n;
         idx += (size_t)gridDim.x * blockDim.x)
        dst[idx] = (bf16)src[idx];
}

__global__ void prep_bias(const float* __restrict__ bi, const float* __restrict__ bh,
                          float* __restrict__ bias2) {
    for (int idx = blockIdx.x * blockDim.x + threadIdx.x; idx < G4H;
         idx += gridDim.x * blockDim.x)
        bias2[idx] = bi[idx] + bh[idx];
}

__global__ void prep_state(bf16* __restrict__ h0, float* __restrict__ c) {
    for (int idx = blockIdx.x * blockDim.x + threadIdx.x; idx < B_ * H_;
         idx += gridDim.x * blockDim.x) {
        h0[idx] = (bf16)0.0f;
        c[idx]  = 0.0f;
    }
}

// ---------------- fragment helpers ----------------
// A fragment (16x32 bf16, row M = lane%16): elems 0..15 -> K = 16*half+e.
// Two 16B loads from a row-major row.
__device__ __forceinline__ v16bf load_a(const bf16* __restrict__ row, int k0, int lh) {
    v8bf lo = *(const v8bf*)(row + k0 + 16 * lh);
    v8bf hi = *(const v8bf*)(row + k0 + 16 * lh + 8);
    v16bf a;
#pragma unroll
    for (int e = 0; e < 8; e++) { a[e] = lo[e]; a[e + 8] = hi[e]; }
    return a;
}

// ---------------- xg = xt @ W_ih.T + bias2 (stored bf16 [S][B][4096]) ----------------
__global__ __launch_bounds__(256) void xg_gemm(const bf16* __restrict__ xtb,
                                               const bf16* __restrict__ Wih,
                                               const float* __restrict__ bias2,
                                               bf16* __restrict__ xg) {
    __shared__ __align__(32) bf16 ldsB[128][32];
    const int tid = threadIdx.x;
    const int wv = tid >> 5, ln = tid & 31, lh = ln >> 4, lm = ln & 15;
    const int m0 = (blockIdx.x >> 5) * 128;   // 512 M-blocks over 65536 rows
    const int n0 = (blockIdx.x & 31) * 128;   // 32 N-blocks over 4096 cols

    v8f acc[8];
#pragma unroll
    for (int j = 0; j < 8; j++)
#pragma unroll
        for (int r = 0; r < 8; r++) acc[j][r] = 0.0f;

    const bf16* arow = xtb + (size_t)(m0 + 16 * wv + lm) * IN_PAD;
    const int jj = tid >> 1, hh = tid & 1;
    const bf16* wsrc = Wih + (size_t)(n0 + jj) * IN_PAD + hh * 16;

    for (int k0 = 0; k0 < IN_PAD; k0 += 32) {
        *(v16bf*)(&ldsB[jj][hh * 16]) = *(const v16bf*)(wsrc + k0);
        __syncthreads();
        v16bf a = load_a(arow, k0, lh);
#pragma unroll
        for (int j = 0; j < 8; j++) {
            v16bf b = *(const v16bf*)(&ldsB[j * 16 + lm][16 * lh]);
            acc[j] = __builtin_amdgcn_wmma_f32_16x16x32_bf16(false, a, false, b,
                                                             (short)0, acc[j], false, false);
        }
        __syncthreads();
    }
#pragma unroll
    for (int j = 0; j < 8; j++) {
        int col = n0 + j * 16 + lm;
        float bb = bias2[col];
#pragma unroll
        for (int r = 0; r < 8; r++) {
            int mrow = m0 + 16 * wv + r + 8 * lh;
            xg[(size_t)mrow * G4H + col] = (bf16)(acc[j][r] + bb);
        }
    }
}

// ---------------- one LSTM timestep ----------------
// g = xg[t] + h @ W_hh.T ; gates i,f,g,o ; c' = f*c + i*tanh(g) ; h' = o*tanh(c')
// Block handles H-columns [n0, n0+32) across all 4 gate sections (128 local cols).
// The whole 128x1024 bf16 W_hh panel (256 KB) is pulled into LDS by ONE 3D
// tensor_load_to_lds (TDM), then the K-loop runs barrier-free out of LDS.
__global__ __launch_bounds__(256) void lstm_step(const bf16* __restrict__ hin,
                                                 bf16* __restrict__ hout,
                                                 float* __restrict__ c,
                                                 const bf16* __restrict__ Whh,
                                                 const bf16* __restrict__ xg,
                                                 int t) {
    __shared__ __align__(128) bf16 ldsB[128][H_];   // 256 KB, LDS offset 0
    const int tid = threadIdx.x;
    const int wv = tid >> 5, ln = tid & 31, lh = ln >> 4, lm = ln & 15;
    const int n0 = blockIdx.x * 32;           // 32 blocks cover H

    // ---- TDM: load 4 gate tiles (32 rows x 1024 K each) as one 3D tile ----
    if (tid < 32) {   // wave 0 issues; EXEC ignored by TDM, one op per wave
        unsigned long long ga =
            (unsigned long long)(uintptr_t)(Whh + (size_t)n0 * H_);
        v4u g0;
        g0[0] = 1u;                                      // count=1 (valid), no gather
        g0[1] = 0u;                                      // lds_addr = 0 (ldsB at base)
        g0[2] = (unsigned)ga;                            // global_addr[31:0]
        g0[3] = ((unsigned)(ga >> 32) & 0x01FFFFFFu)     // global_addr[56:32]
                | 0x80000000u;                           // type=2 ("image")
        v8i g1;
        g1[0] = 0x00010000;                  // wg_mask=0, data_size=1 (2B)
        g1[1] = (int)(H_ << 16);             // tensor_dim0 = 1024 (lo16 in [31:16])
        g1[2] = (int)(G4H << 16);            // dim0 hi=0 | tensor_dim1 = 4096 (lo16)
        g1[3] = (int)(H_ << 16);             // dim1 hi=0 | tile_dim0 = 1024
        g1[4] = (4 << 16) | 32;              // tile_dim1 = 32, tile_dim2 = 4 gates
        g1[5] = H_;                          // tensor_dim0_stride = 1024 (lo32)
        g1[6] = 0;                           // stride0 hi | dim1_stride lo16 (=0)
        g1[7] = (int)((H_ * (unsigned)H_) >> 16);  // dim1_stride = H*H (>>16) = 16
        v4i g2;
        g2[0] = 4;                           // tensor_dim2 = 4 (gate planes)
        g2[1] = 0; g2[2] = 0; g2[3] = 0;
        v4i g3;
        g3[0] = 0; g3[1] = 0; g3[2] = 0; g3[3] = 0;
        v8i g4;                              // extra group (clang-23 6-arg form);
        g4[0] = 0; g4[1] = 0; g4[2] = 0; g4[3] = 0;   // zero-filled = inert
        g4[4] = 0; g4[5] = 0; g4[6] = 0; g4[7] = 0;
        __builtin_amdgcn_tensor_load_to_lds(g0, g1, g2, g3, g4, 0);
        __builtin_amdgcn_s_wait_tensorcnt(0);
    }
    __syncthreads();

    v8f acc[8];
#pragma unroll
    for (int j = 0; j < 8; j++)
#pragma unroll
        for (int r = 0; r < 8; r++) acc[j][r] = 0.0f;

    const bf16* arow = hin + (size_t)(16 * wv + lm) * H_;

    for (int k0 = 0; k0 < H_; k0 += 32) {    // barrier-free hot loop
        v16bf a = load_a(arow, k0, lh);
#pragma unroll
        for (int j = 0; j < 8; j++) {
            v16bf b = *(const v16bf*)(&ldsB[j * 16 + lm][k0 + 16 * lh]);
            acc[j] = __builtin_amdgcn_wmma_f32_16x16x32_bf16(false, a, false, b,
                                                             (short)0, acc[j], false, false);
        }
    }

    const bf16* xgt = xg + (size_t)t * B_ * G4H;
#pragma unroll
    for (int jp = 0; jp < 2; jp++) {
#pragma unroll
        for (int r = 0; r < 8; r++) {
            int m = 16 * wv + r + 8 * lh;
            int n = n0 + 16 * jp + lm;
            size_t xb = (size_t)m * G4H + n;
            float gi = acc[0 + jp][r] + (float)xgt[xb];
            float gf = acc[2 + jp][r] + (float)xgt[xb + H_];
            float gg = acc[4 + jp][r] + (float)xgt[xb + 2 * H_];
            float go = acc[6 + jp][r] + (float)xgt[xb + 3 * H_];
            float iv = sigmoidf_(gi), fv = sigmoidf_(gf);
            float gv = tanhf(gg),     ov = sigmoidf_(go);
            size_t ci = (size_t)m * H_ + n;
            float cn = fv * c[ci] + iv * gv;
            c[ci] = cn;
            hout[ci] = (bf16)(ov * tanhf(cn));
        }
    }
}

// ---------------- final FC: out = h @ W_fc.T + b_fc  [128 x 64] fp32 ----------------
__global__ __launch_bounds__(256) void fc_final(const bf16* __restrict__ h,
                                                const bf16* __restrict__ Wfc,
                                                const float* __restrict__ bfc,
                                                float* __restrict__ out) {
    __shared__ __align__(32) bf16 ldsB[64][32];
    const int tid = threadIdx.x;
    const int wv = tid >> 5, ln = tid & 31, lh = ln >> 4, lm = ln & 15;

    v8f acc[4];
#pragma unroll
    for (int j = 0; j < 4; j++)
#pragma unroll
        for (int r = 0; r < 8; r++) acc[j][r] = 0.0f;

    const bf16* arow = h + (size_t)(16 * wv + lm) * H_;
    for (int k0 = 0; k0 < H_; k0 += 32) {
        if (tid < 128) {
            int jj = tid >> 1, hh = tid & 1;
            *(v16bf*)(&ldsB[jj][hh * 16]) =
                *(const v16bf*)(Wfc + (size_t)jj * H_ + k0 + hh * 16);
        }
        __syncthreads();
        v16bf a = load_a(arow, k0, lh);
#pragma unroll
        for (int j = 0; j < 4; j++) {
            v16bf b = *(const v16bf*)(&ldsB[j * 16 + lm][16 * lh]);
            acc[j] = __builtin_amdgcn_wmma_f32_16x16x32_bf16(false, a, false, b,
                                                             (short)0, acc[j], false, false);
        }
        __syncthreads();
    }
#pragma unroll
    for (int j = 0; j < 4; j++) {
        int n = j * 16 + lm;
        float bb = bfc[n];
#pragma unroll
        for (int r = 0; r < 8; r++) {
            int m = 16 * wv + r + 8 * lh;
            out[(size_t)m * O_ + n] = acc[j][r] + bb;
        }
    }
}

// ---------------- launcher ----------------
extern "C" void kernel_launch(void* const* d_in, const int* in_sizes, int n_in,
                              void* d_out, int out_size, void* d_ws, size_t ws_size,
                              hipStream_t stream) {
    (void)in_sizes; (void)n_in; (void)out_size; (void)ws_size;
    const float* x    = (const float*)d_in[0];
    const float* mask = (const float*)d_in[1];
    const float* ti   = (const float*)d_in[2];
    const float* W_ih = (const float*)d_in[3];
    const float* W_hh = (const float*)d_in[4];
    const float* b_ih = (const float*)d_in[5];
    const float* b_hh = (const float*)d_in[6];
    const float* W_fc = (const float*)d_in[7];
    const float* b_fc = (const float*)d_in[8];
    float* out = (float*)d_out;

    char* ws = (char*)d_ws;
    auto alloc = [&](size_t bytes) -> char* {
        char* p = ws;
        ws += (bytes + 255) & ~(size_t)255;
        return p;
    };
    bf16*  xtb   = (bf16*)alloc((size_t)S_ * B_ * IN_PAD * 2);
    bf16*  Wihb  = (bf16*)alloc((size_t)G4H * IN_PAD * 2);
    bf16*  Whhb  = (bf16*)alloc((size_t)G4H * H_ * 2);
    bf16*  Wfcb  = (bf16*)alloc((size_t)O_ * H_ * 2);
    float* bias2 = (float*)alloc((size_t)G4H * 4);
    bf16*  xg    = (bf16*)alloc((size_t)S_ * B_ * G4H * 2);
    bf16*  h0    = (bf16*)alloc((size_t)B_ * H_ * 2);
    bf16*  h1    = (bf16*)alloc((size_t)B_ * H_ * 2);
    float* cbuf  = (float*)alloc((size_t)B_ * H_ * 4);

    prep_xtb <<<2048, 256, 0, stream>>>(x, mask, ti, xtb);
    prep_wih <<<1024, 256, 0, stream>>>(W_ih, Wihb);
    prep_f2b <<<2048, 256, 0, stream>>>(W_hh, Whhb, (size_t)G4H * H_);
    prep_f2b <<<64,   256, 0, stream>>>(W_fc, Wfcb, (size_t)O_ * H_);
    prep_bias<<<16,   256, 0, stream>>>(b_ih, b_hh, bias2);
    prep_state<<<256, 256, 0, stream>>>(h0, cbuf);

    xg_gemm<<<16384, 256, 0, stream>>>(xtb, Wihb, bias2, xg);

    bf16* hb[2] = { h0, h1 };
    for (int t = 0; t < S_; t++)
        lstm_step<<<32, 256, 0, stream>>>(hb[t & 1], hb[(t + 1) & 1], cbuf, Whhb, xg, t);

    fc_final<<<1, 256, 0, stream>>>(h0, Wfcb, b_fc, out);
}
